// MHSA_block_52218212385434
// MI455X (gfx1250) — matrix operand order repaired and checked by reading, not compiled, not run on hardware
//
#include <hip/hip_runtime.h>
#include <hip/hip_bf16.h>

typedef __attribute__((ext_vector_type(16))) _Float16 v16h;
typedef __attribute__((ext_vector_type(8)))  _Float16 v8h;
typedef __attribute__((ext_vector_type(8)))  float    v8f;
typedef __attribute__((ext_vector_type(4)))  float    v4f;
typedef int vb128 __attribute__((vector_size(16)));   // builtin's pointee type

constexpr int B_  = 2;
constexpr int H_  = 8;
constexpr int P_  = 2048;   // sequence
constexpr int D_  = 128;    // d_model
constexpr int DF_ = 32;     // per-head dim
constexpr int C_  = H_ * DF_;  // 256 concat-head dim
constexpr float INV_SQRT_DF = 0.17677669529663687f;  // 1/sqrt(32)

// ---- WMMA wrapper: D = A(16xK=32, f16) * B(32x16, f16) + C(16x16 f32) ----
__device__ __forceinline__ v8f wmma_f16(v16h a, v16h b, v8f c) {
    return __builtin_amdgcn_wmma_f32_16x16x32_f16(false, a, false, b,
                                                  (short)0, c, false, false);
}

// ---- async global -> LDS copy (16B), tracked by ASYNCcnt ----
__device__ __forceinline__ void async_copy_b128(const _Float16* gsrc, _Float16* ldst) {
#if __has_builtin(__builtin_amdgcn_global_load_async_to_lds_b128)
    __builtin_amdgcn_global_load_async_to_lds_b128((vb128*)gsrc, (vb128*)ldst, 0, 0);
#else
    asm volatile("global_load_async_to_lds_b128 %0, %1, off"
                 :: "v"((uint32_t)(uintptr_t)ldst), "v"(gsrc)
                 : "memory");
#endif
}
__device__ __forceinline__ void wait_async0() {
#if __has_builtin(__builtin_amdgcn_s_wait_asynccnt)
    __builtin_amdgcn_s_wait_asynccnt(0);
#else
    asm volatile("s_wait_asynccnt 0" ::: "memory");
#endif
}

// A fragment (16-bit A 16x32 layout): lane holds row M=(lane&15); lane-half
// selects K range: VGPR0-7 = K kc+2*ah+{0..15} (contiguous),
// ah = (lane>>4)*8.  `row` already points at this lane's row.
__device__ __forceinline__ v16h frag_a_h(const _Float16* row, int kc, int ah) {
    v8h lo = *(const v8h*)(row + kc + 2 * ah);
    v8h hh = *(const v8h*)(row + kc + 2 * ah + 8);
    v16h r;
#pragma unroll
    for (int j = 0; j < 8; ++j) { r[j] = lo[j]; r[8 + j] = hh[j]; }
    return r;
}
__device__ __forceinline__ v16h frag_a_f32(const float* __restrict__ row, int kc, int ah) {
    const float4* p0 = (const float4*)(row + kc + 2 * ah);
    const float4* p1 = (const float4*)(row + kc + 2 * ah + 8);
    float4 a = p0[0], b = p0[1], c = p1[0], d = p1[1];
    v16h r;
    r[0]=(_Float16)a.x;  r[1]=(_Float16)a.y;  r[2]=(_Float16)a.z;  r[3]=(_Float16)a.w;
    r[4]=(_Float16)b.x;  r[5]=(_Float16)b.y;  r[6]=(_Float16)b.z;  r[7]=(_Float16)b.w;
    r[8]=(_Float16)c.x;  r[9]=(_Float16)c.y;  r[10]=(_Float16)c.z; r[11]=(_Float16)c.w;
    r[12]=(_Float16)d.x; r[13]=(_Float16)d.y; r[14]=(_Float16)d.z; r[15]=(_Float16)d.w;
    return r;
}
// B fragment (16-bit B 32x16 layout): lane holds column N=(lane&15); lane-half
// selects K range: half0 K=kc+0..15, half1 K=kc+16..31, packed 2/VGPR
// sequentially.  Pass kbase = kc + (lane>>4)*16.
__device__ __forceinline__ v16h frag_b_h(const _Float16* __restrict__ row, int kbase) {
    v8h lo = *(const v8h*)(row + kbase);
    v8h hh = *(const v8h*)(row + kbase + 8);
    v16h r;
#pragma unroll
    for (int j = 0; j < 8; ++j) { r[j] = lo[j]; r[8 + j] = hh[j]; }
    return r;
}
__device__ __forceinline__ v16h frag_b_f32(const float* __restrict__ row, int kbase) {
    const float4* p = (const float4*)(row + kbase);
    float4 a = p[0], b = p[1], c = p[2], d = p[3];
    v16h r;
    r[0]=(_Float16)a.x;  r[1]=(_Float16)a.y;  r[2]=(_Float16)a.z;  r[3]=(_Float16)a.w;
    r[4]=(_Float16)b.x;  r[5]=(_Float16)b.y;  r[6]=(_Float16)b.z;  r[7]=(_Float16)b.w;
    r[8]=(_Float16)c.x;  r[9]=(_Float16)c.y;  r[10]=(_Float16)c.z; r[11]=(_Float16)c.w;
    r[12]=(_Float16)d.x; r[13]=(_Float16)d.y; r[14]=(_Float16)d.z; r[15]=(_Float16)d.w;
    return r;
}

// ============================================================================
// Kernel A: qkv = x @ Wqkv^T  (M=B*P=4096, N=768, K=128), WMMA-tiled.
// Scatters q -> Qtilde (Wtalk/sqrt(DF) absorbed, per output head g),
// k -> Ktilde [B,P,256], v -> V^T [B,H,32,P].
// ============================================================================
__global__ void __launch_bounds__(256)
qkv_kernel(const float* __restrict__ x, const float* __restrict__ Wqkv,
           const float* __restrict__ Wtalk,
           _Float16* __restrict__ qt, _Float16* __restrict__ kt,
           _Float16* __restrict__ vt)
{
    int gid  = blockIdx.x * (blockDim.x >> 5) + (threadIdx.x >> 5);
    int lane = threadIdx.x & 31;
    int mt = gid / 48, nt = gid % 48;           // 256 x 48 tiles
    int m0 = mt * 16, n0 = nt * 16;
    int nloc = lane & 15;
    int hi   = lane >> 4;
    int ah   = hi * 8;

    const float* arow = x    + (size_t)(m0 + nloc) * D_;
    const float* brow = Wqkv + (size_t)(n0 + nloc) * D_;

    v8f acc = {};
#pragma unroll
    for (int c = 0; c < 4; ++c) {
        v16h af = frag_a_f32(arow, c * 32, ah);
        v16h bf = frag_b_f32(brow, c * 32 + hi * 16);
        acc = wmma_f16(af, bf, acc);
    }

    int n    = n0 + nloc;
    int part = n >> 8;          // 0=q 1=k 2=v (uniform per wave)
    int h    = (n >> 5) & 7;    // head (uniform per wave)
    int dfc  = n & 31;

    if (part == 0) {
        float wt[8];
#pragma unroll
        for (int g = 0; g < 8; ++g) wt[g] = Wtalk[g * 8 + h] * INV_SQRT_DF;
#pragma unroll
        for (int r = 0; r < 8; ++r) {
            int mrow = m0 + ah + r;                  // M = r + 8*hi
            int b = mrow >> 11, p = mrow & (P_ - 1);
#pragma unroll
            for (int g = 0; g < 8; ++g)
                qt[((size_t)(b * H_ + g) * P_ + p) * C_ + h * DF_ + dfc] =
                    (_Float16)(acc[r] * wt[g]);
        }
    } else if (part == 1) {
#pragma unroll
        for (int r = 0; r < 8; ++r) {
            int mrow = m0 + ah + r;
            int b = mrow >> 11, p = mrow & (P_ - 1);
            kt[((size_t)b * P_ + p) * C_ + h * DF_ + dfc] = (_Float16)acc[r];
        }
    } else {
#pragma unroll
        for (int r = 0; r < 8; ++r) {
            int mrow = m0 + ah + r;
            int b = mrow >> 11, p = mrow & (P_ - 1);
            vt[((size_t)(b * H_ + h) * DF_ + dfc) * P_ + p] = (_Float16)acc[r];
        }
    }
}

// ============================================================================
// Kernel B: fused flash attention (transposed scores) per (b, g, 16-query tile)
//   S^T = Ktilde * Qtilde_g^T  (K-dim 256) ; +bias(NT) ; online softmax
//   O^T += V^T * P^T
// One wave per task; 8 waves/block share (b,g); K-tiles staged to LDS via
// async global->LDS loads, double buffered, prefetch overlapped with WMMA.
// ============================================================================
__global__ void __launch_bounds__(256)
attn_kernel(const float* __restrict__ bias,
            const _Float16* __restrict__ qt, const _Float16* __restrict__ kt,
            const _Float16* __restrict__ vt, _Float16* __restrict__ oattn)
{
    __shared__ __align__(16) _Float16 ksm[2][32 * C_];   // 2 x 16 KiB K-chunk buffers

    int tid  = threadIdx.x;
    int task = blockIdx.x * (blockDim.x >> 5) + (tid >> 5);
    int lane = tid & 31;
    int b  = task >> 10;          // uniform per block
    int g  = (task >> 7) & 7;     // uniform per block
    int q0 = (task & 127) * 16;
    int qloc = lane & 15;
    int hi   = lane >> 4;
    int ah   = hi * 8;

    // Qtilde B-fragments for all 8 K-chunks (held for the whole key loop)
    const _Float16* qrow = qt + ((size_t)(b * H_ + g) * P_ + q0 + qloc) * C_;
    v16h qf[8];
#pragma unroll
    for (int c = 0; c < 8; ++c) qf[c] = frag_b_h(qrow, c * 32 + hi * 16);

    v8f o0 = {}, o1 = {};                 // O^T tiles: df 0-15 / 16-31, N=queries
    float m_run = -INFINITY, l_run = 0.f;

    const float*    brow  = bias + ((size_t)g * P_ + q0 + qloc) * P_;
    const _Float16* kbase = kt + (size_t)b * P_ * C_;
    const _Float16* vrow0 = vt + ((size_t)(b * H_ + g) * DF_ + 0  + qloc) * P_;
    const _Float16* vrow1 = vt + ((size_t)(b * H_ + g) * DF_ + 16 + qloc) * P_;

    // cooperative staging: 256 threads x 32 f16 = one 32x256 K-chunk
    const _Float16* ksrc = kbase + tid * 32;
    _Float16* kdst0 = &ksm[0][tid * 32];
    _Float16* kdst1 = &ksm[1][tid * 32];

    // prologue: stage chunk 0 into buffer 0
#pragma unroll
    for (int j = 0; j < 4; ++j) async_copy_b128(ksrc + j * 8, kdst0 + j * 8);

#pragma unroll 1
    for (int i = 0; i < P_ / 32; ++i) {
        int cur = i & 1;
        int k0  = i * 32;

        wait_async0();        // my slice of buffer `cur` has landed
        __syncthreads();      // everyone's slice landed; prior reads of other buf done

        if (i + 1 < P_ / 32) {   // prefetch next chunk into the other buffer
            const _Float16* src = ksrc + (size_t)(i + 1) * 32 * C_;
            _Float16* dst = cur ? kdst0 : kdst1;
#pragma unroll
            for (int j = 0; j < 4; ++j) async_copy_b128(src + j * 8, dst + j * 8);
        }

        const _Float16* krow0 = &ksm[cur][(size_t)qloc * C_];        // keys k0..k0+15
        const _Float16* krow1 = &ksm[cur][(size_t)(16 + qloc) * C_]; // keys k0+16..+31
        v8f s0 = {}, s1 = {};
#pragma unroll
        for (int c = 0; c < 8; ++c) s0 = wmma_f16(frag_a_h(krow0, c * 32, ah), qf[c], s0);
#pragma unroll
        for (int c = 0; c < 8; ++c) s1 = wmma_f16(frag_a_h(krow1, c * 32, ah), qf[c], s1);

        // bias (streamed once -> non-temporal): query q0+qloc, keys k0+ah+{0..7}/+16
        const v4f* bp0 = (const v4f*)(brow + k0 + ah);
        const v4f* bp1 = (const v4f*)(brow + k0 + 16 + ah);
        v4f b0a = __builtin_nontemporal_load(bp0);
        v4f b0b = __builtin_nontemporal_load(bp0 + 1);
        v4f b1a = __builtin_nontemporal_load(bp1);
        v4f b1b = __builtin_nontemporal_load(bp1 + 1);
#pragma unroll
        for (int j = 0; j < 4; ++j) { s0[j] += b0a[j]; s0[4 + j] += b0b[j];
                                      s1[j] += b1a[j]; s1[4 + j] += b1b[j]; }

        // online softmax: per-query stats live in lane pair (l, l^16)
        float tmax = s0[0];
#pragma unroll
        for (int j = 1; j < 8; ++j) tmax = fmaxf(tmax, s0[j]);
#pragma unroll
        for (int j = 0; j < 8; ++j) tmax = fmaxf(tmax, s1[j]);
        tmax = fmaxf(tmax, __shfl_xor(tmax, 16, 32));
        float nm = fmaxf(m_run, tmax);
        float ex = __expf(m_run - nm);

        float plo[8], phi[8];
        float lsum = 0.f;
#pragma unroll
        for (int j = 0; j < 8; ++j) { plo[j] = __expf(s0[j] - nm); lsum += plo[j]; }
#pragma unroll
        for (int j = 0; j < 8; ++j) { phi[j] = __expf(s1[j] - nm); lsum += phi[j]; }
        lsum += __shfl_xor(lsum, 16, 32);
        l_run = l_run * ex + lsum;
        m_run = nm;
#pragma unroll
        for (int j = 0; j < 8; ++j) { o0[j] *= ex; o1[j] *= ex; }

        // Build P^T B-fragment (32 keys x 16 queries): one cross-half shuffle
        float olo[8], ohi[8];
#pragma unroll
        for (int j = 0; j < 8; ++j) { olo[j] = __shfl_xor(plo[j], 16, 32);
                                      ohi[j] = __shfl_xor(phi[j], 16, 32); }
        bool ishi = (hi != 0);
        v16h pf;
#pragma unroll
        for (int j = 0; j < 8; ++j) pf[j]     = (_Float16)(ishi ? ohi[j] : plo[j]);
#pragma unroll
        for (int j = 0; j < 8; ++j) pf[8 + j] = (_Float16)(ishi ? phi[j] : olo[j]);

        // O^T += V^T * P^T
        v16h v0 = frag_a_h(vrow0, k0, ah);
        v16h v1 = frag_a_h(vrow1, k0, ah);
        o0 = wmma_f16(v0, pf, o0);
        o1 = wmma_f16(v1, pf, o1);
    }

    float linv = 1.f / l_run;
    // store O[b, p=q, g*32+df]  (O^T layout: df = r + ah (+16), q = qloc)
    _Float16* obase = oattn + ((size_t)b * P_ + q0) * C_ + g * DF_;
#pragma unroll
    for (int r = 0; r < 8; ++r) {
        obase[(size_t)qloc * C_ + r + ah]      = (_Float16)(o0[r] * linv);
        obase[(size_t)qloc * C_ + 16 + r + ah] = (_Float16)(o1[r] * linv);
    }
}

// ============================================================================
// Kernel C: out = O[B*P,256] @ Wproj^T  (M=4096, N=128, K=256), fp32 stores.
// ============================================================================
__global__ void __launch_bounds__(256)
proj_kernel(const _Float16* __restrict__ oattn, const float* __restrict__ Wproj,
            float* __restrict__ out)
{
    int gid  = blockIdx.x * (blockDim.x >> 5) + (threadIdx.x >> 5);
    int lane = threadIdx.x & 31;
    int mt = gid >> 3, nt = gid & 7;            // 256 x 8 tiles
    int m0 = mt * 16, n0 = nt * 16;
    int nloc = lane & 15;
    int hi   = lane >> 4;
    int ah   = hi * 8;

    const _Float16* arow = oattn + (size_t)(m0 + nloc) * C_;
    const float*    brow = Wproj + (size_t)(n0 + nloc) * C_;

    v8f acc = {};
#pragma unroll
    for (int c = 0; c < 8; ++c)
        acc = wmma_f16(frag_a_h(arow, c * 32, ah), frag_b_f32(brow, c * 32 + hi * 16), acc);

#pragma unroll
    for (int r = 0; r < 8; ++r)
        out[(size_t)(m0 + ah + r) * D_ + n0 + nloc] = acc[r];
}

// ============================================================================
extern "C" void kernel_launch(void* const* d_in, const int* in_sizes, int n_in,
                              void* d_out, int out_size, void* d_ws, size_t ws_size,
                              hipStream_t stream)
{
    (void)in_sizes; (void)n_in; (void)out_size; (void)ws_size;
    const float* x     = (const float*)d_in[0];
    const float* bias  = (const float*)d_in[1];
    const float* Wqkv  = (const float*)d_in[2];
    const float* Wtalk = (const float*)d_in[3];
    const float* Wproj = (const float*)d_in[4];
    float* out = (float*)d_out;

    char* ws = (char*)d_ws;
    const size_t QT_BYTES = (size_t)B_ * H_ * P_ * C_ * sizeof(_Float16);  // 16 MiB
    const size_t KT_BYTES = (size_t)B_ * P_ * C_ * sizeof(_Float16);       //  2 MiB
    const size_t VT_BYTES = (size_t)B_ * H_ * DF_ * P_ * sizeof(_Float16); //  2 MiB
    _Float16* qt = (_Float16*)ws;
    _Float16* kt = (_Float16*)(ws + QT_BYTES);
    _Float16* vt = (_Float16*)(ws + QT_BYTES + KT_BYTES);
    _Float16* oa = (_Float16*)(ws + QT_BYTES + KT_BYTES + VT_BYTES);

    // A: 256*48 = 12288 wave-tiles, 8 waves/block
    qkv_kernel<<<1536, 256, 0, stream>>>(x, Wqkv, Wtalk, qt, kt, vt);
    // B: 2*8*128 = 2048 wave-tasks
    attn_kernel<<<256, 256, 0, stream>>>(bias, qt, kt, vt, oa);
    // C: 256*8 = 2048 wave-tiles
    proj_kernel<<<256, 256, 0, stream>>>(oa, Wproj, out);
}
